// RoFSigLipLoss_13761075216803
// MI455X (gfx1250) — matrix operand order repaired and checked by reading, not compiled, not run on hardware
//
#include <hip/hip_runtime.h>

#define T_CONST      0.9f
#define TAU_CONST    0.1f
#define BETA_CONST   0.25f
#define LAM_CONST    0.01f
#define MARGIN_CONST 0.2f
#define W_DCM        0.2f
#define W_DTM        0.2f
#define W_DENSE      0.2f

#define B_ROWS  2048
#define D_DIM   1024
#define NG_COLS 16384
#define P_PARTS 16      // column-partials per row for ng/emb stats (gridDim.y * 2)

typedef _Float16 v16h __attribute__((ext_vector_type(16)));
typedef _Float16 v8h  __attribute__((ext_vector_type(8)));
typedef float    v8f  __attribute__((ext_vector_type(8)));

union V16U { v16h v; v8h h[2]; };

// ---------------------------------------------------------------- utilities

__device__ __forceinline__ float softplus_f(float z) {
  // -log_sigmoid(-z) = softplus(z), numerically stable
  return fmaxf(z, 0.0f) + log1pf(expf(-fabsf(z)));
}

__device__ __forceinline__ float wave_reduce_add(float v) {
  #pragma unroll
  for (int m = 1; m < 32; m <<= 1) v += __shfl_xor(v, m, 32);
  return v;
}
__device__ __forceinline__ float half_reduce_add(float v) {
  #pragma unroll
  for (int m = 1; m < 16; m <<= 1) v += __shfl_xor(v, m, 16);
  return v;
}
__device__ __forceinline__ float half_reduce_max(float v) {
  #pragma unroll
  for (int m = 1; m < 16; m <<= 1) v = fmaxf(v, __shfl_xor(v, m, 16));
  return v;
}

// -------- JAX threefry2x32 with key = (0, 42) (jax.random.key(42)) --------

__device__ __forceinline__ unsigned rotl32(unsigned x, int d) {
  return (x << d) | (x >> (32 - d));
}

__device__ __forceinline__ void threefry2x32_0_42(unsigned c0, unsigned c1,
                                                  unsigned& o0, unsigned& o1) {
  const unsigned k0 = 0u, k1 = 42u;
  const unsigned k2 = k0 ^ k1 ^ 0x1BD11BDAu;
  unsigned x0 = c0 + k0, x1 = c1 + k1;
#define TF_ROUND(r) { x0 += x1; x1 = rotl32(x1, (r)); x1 ^= x0; }
  TF_ROUND(13) TF_ROUND(15) TF_ROUND(26) TF_ROUND(6)  x0 += k1; x1 += k2 + 1u;
  TF_ROUND(17) TF_ROUND(29) TF_ROUND(16) TF_ROUND(24) x0 += k2; x1 += k0 + 2u;
  TF_ROUND(13) TF_ROUND(15) TF_ROUND(26) TF_ROUND(6)  x0 += k0; x1 += k1 + 3u;
  TF_ROUND(17) TF_ROUND(29) TF_ROUND(16) TF_ROUND(24) x0 += k1; x1 += k2 + 4u;
  TF_ROUND(13) TF_ROUND(15) TF_ROUND(26) TF_ROUND(6)  x0 += k2; x1 += k0 + 5u;
#undef TF_ROUND
  o0 = x0; o1 = x1;
}

__device__ __forceinline__ float erfinv_f(float x) {  // Giles (2012) approximation
  float w = -logf((1.0f - x) * (1.0f + x));
  float p;
  if (w < 5.0f) {
    w = w - 2.5f;
    p = 2.81022636e-08f;
    p = fmaf(p, w, 3.43273939e-07f);
    p = fmaf(p, w, -3.5233877e-06f);
    p = fmaf(p, w, -4.39150654e-06f);
    p = fmaf(p, w, 0.00021858087f);
    p = fmaf(p, w, -0.00125372503f);
    p = fmaf(p, w, -0.00417768164f);
    p = fmaf(p, w, 0.246640727f);
    p = fmaf(p, w, 1.50140941f);
  } else {
    w = sqrtf(w) - 3.0f;
    p = -0.000200214257f;
    p = fmaf(p, w, 0.000100950558f);
    p = fmaf(p, w, 0.00134934322f);
    p = fmaf(p, w, -0.00367342844f);
    p = fmaf(p, w, 0.00573950773f);
    p = fmaf(p, w, -0.0076224613f);
    p = fmaf(p, w, 0.00943887047f);
    p = fmaf(p, w, 1.00167406f);
    p = fmaf(p, w, 2.83297682f);
  }
  return p * x;
}

__device__ __forceinline__ float bits_to_normal(unsigned bits) {
  // jax.random.normal: bits -> uniform(nextafter(-1,0), 1) -> sqrt(2)*erfinv(u)
  const float u01 = __uint_as_float((bits >> 9) | 0x3F800000u) - 1.0f;
  const float lo = -0.99999994f;
  float u = u01 * (1.0f - lo) + lo;
  u = fmaxf(lo, u);
  return 1.41421356237309515f * erfinv_f(u);
}

// ---------------------------------------------------------------- kernels

__global__ void init_accum_k(float* a) {
  if (threadIdx.x < 64) a[threadIdx.x] = 0.0f;
}

// f32 -> (f16 hi, f16 lo) split-precision planes
__global__ void split_f16_k(const float* __restrict__ x,
                            _Float16* __restrict__ hi,
                            _Float16* __restrict__ lo, size_t n) {
  size_t i = (size_t)blockIdx.x * blockDim.x + threadIdx.x;
  const size_t stride = (size_t)gridDim.x * blockDim.x;
  for (; i < n; i += stride) {
    const float v = x[i];
    const _Float16 h = (_Float16)v;
    hi[i] = h;
    lo[i] = (_Float16)(v - (float)h);
  }
}

// out[row] = scale * dot(a[row], b[row])  -- exact f32 diagonals
__global__ void row_dot_k(const float* __restrict__ a, const float* __restrict__ b,
                          const float* __restrict__ scaleP, float* __restrict__ out) {
  const int w = threadIdx.x >> 5, lane = threadIdx.x & 31;
  const int row = blockIdx.x * 8 + w;
  const float* pa = a + (size_t)row * D_DIM;
  const float* pb = b + (size_t)row * D_DIM;
  float s = 0.0f;
  for (int k = lane; k < D_DIM; k += 32) s = fmaf(pa[k], pb[k], s);
  s = wave_reduce_add(s);
  if (lane == 0) out[row] = (*scaleP) * s;
}

// Fused split-precision WMMA GEMM: C = scale * X @ Y^T (+ fused reductions)
//  mode 1: gt siglip     (softplus with diag labels -> accum[slot])
//  mode 2: dense siglip  (same)
//  mode 3: ng            (softplus(+z) sum -> accum[slot]; row max/E1/E2 partials)
//  mode 4: emb           (row max/E1/E2 partials only)
// Block: 256 threads = 8 waves as 4 row-waves x 2 col-waves over a 64x32 tile.
// X block (64 rows x 1024 K, hi+lo f16) staged in 256 KB of dynamic LDS.
__global__ void __launch_bounds__(256)
gemm_fused_k(const _Float16* __restrict__ Xhi, const _Float16* __restrict__ Xlo,
             const _Float16* __restrict__ Yhi, const _Float16* __restrict__ Ylo,
             const float* __restrict__ scaleP, const float* __restrict__ biasP,
             int colsPerBlock, int mode,
             float* __restrict__ mOut, float* __restrict__ e1Out,
             float* __restrict__ e2Out,
             float* __restrict__ accum, int accSlot) {
  extern __shared__ _Float16 smem[];
  _Float16* sAhi = smem;
  _Float16* sAlo = smem + 64 * D_DIM;

  const int tid  = threadIdx.x;
  const int lane = tid & 31;
  const int w    = tid >> 5;   // wave 0..7
  const int wr   = w >> 1;     // row-wave 0..3
  const int wc   = w & 1;      // col-wave 0..1
  const int h    = lane >> 4;  // half of wave
  const int lm   = lane & 15;

  const int row64    = blockIdx.x * 64;
  const int colStart = blockIdx.y * colsPerBlock;
  const float scale  = *scaleP;
  const float bias   = *biasP;
  const float invT   = 1.0f / T_CONST;

  // Stage X rows (contiguous 64*1024 elements) into LDS, hi and lo planes.
  {
    const _Float16* gHi = Xhi + (size_t)row64 * D_DIM;
    const _Float16* gLo = Xlo + (size_t)row64 * D_DIM;
    for (int i = tid; i < (64 * D_DIM) / 8; i += 256) {
      *(v8h*)&sAhi[i * 8] = *(const v8h*)&gHi[i * 8];
      *(v8h*)&sAlo[i * 8] = *(const v8h*)&gLo[i * 8];
    }
  }
  __syncthreads();

  float lsum = 0.0f;
  float mrun[8], e1run[8], e2run[8];
  #pragma unroll
  for (int r = 0; r < 8; ++r) { mrun[r] = -INFINITY; e1run[r] = 0.0f; e2run[r] = 0.0f; }

  const int nTiles = colsPerBlock >> 5;            // 32 cols per tile (16 per col-wave)
  const int aBase  = (wr * 16 + lm) * D_DIM + h * 8;  // A row base within LDS block

  #pragma unroll 1
  for (int ct = 0; ct < nTiles; ++ct) {
    const int colBase = colStart + ct * 32 + wc * 16;
    v8f c = {0.f, 0.f, 0.f, 0.f, 0.f, 0.f, 0.f, 0.f};
    const size_t browBase = (size_t)(colBase + lm) * D_DIM + (size_t)(h * 8);

    // Limited unroll: keeps fragments short-lived (no cross-tile hoist -> no spills).
    #pragma unroll 2
    for (int kk = 0; kk < D_DIM; kk += 32) {
      V16U ah, al, bh, bl;
      // A fragment (16-bit 16x32): lanes 0-15 hold K 0..7 / 16..23, lanes 16-31 K 8..15 / 24..31
      const int ka = aBase + kk;
      ah.h[0] = *(const v8h*)&sAhi[ka];
      ah.h[1] = *(const v8h*)&sAhi[ka + 16];
      al.h[0] = *(const v8h*)&sAlo[ka];
      al.h[1] = *(const v8h*)&sAlo[ka + 16];
      // B fragment (32x16): lanes 0-15 hold K 0..7 / 16..23, lanes 16-31 K 8..15 / 24..31
      const size_t kb = browBase + (size_t)kk;
      bh.h[0] = *(const v8h*)&Yhi[kb];
      bh.h[1] = *(const v8h*)&Yhi[kb + 16];
      bl.h[0] = *(const v8h*)&Ylo[kb];
      bl.h[1] = *(const v8h*)&Ylo[kb + 16];
      // split-precision: hi*hi + hi*lo + lo*hi, f32 accumulate
      c = __builtin_amdgcn_wmma_f32_16x16x32_f16(false, ah.v, false, bh.v, (short)0, c, false, false);
      c = __builtin_amdgcn_wmma_f32_16x16x32_f16(false, ah.v, false, bl.v, (short)0, c, false, false);
      c = __builtin_amdgcn_wmma_f32_16x16x32_f16(false, al.v, false, bh.v, (short)0, c, false, false);
    }

    if (mode <= 2) {
      const int gCol = colBase + lm;
      #pragma unroll
      for (int r = 0; r < 8; ++r) {
        const int gRow = row64 + wr * 16 + h * 8 + r;
        const float z = scale * c[r] + bias;
        lsum += (gRow == gCol) ? softplus_f(-z) : softplus_f(z);
      }
    } else {
      #pragma unroll
      for (int r = 0; r < 8; ++r) {
        const float v = scale * c[r];
        if (mode == 3) lsum += softplus_f(v + bias);
        // online row stats over the 16 columns of this half-wave
        const float tm  = half_reduce_max(v);
        const float e   = expf((v - tm) * invT);
        const float e1t = half_reduce_add(e);
        const float e2t = half_reduce_add(e * e);
        const float nm = fmaxf(mrun[r], tm);
        const float s1 = expf((mrun[r] - nm) * invT);
        const float s2 = expf((tm - nm) * invT);
        e1run[r] = e1run[r] * s1 + e1t * s2;
        e2run[r] = e2run[r] * (s1 * s1) + e2t * (s2 * s2);
        mrun[r]  = nm;
      }
    }
  }

  if (mode >= 3 && lm == 0) {
    const int p = blockIdx.y * 2 + wc;
    #pragma unroll
    for (int r = 0; r < 8; ++r) {
      const int gRow = row64 + wr * 16 + h * 8 + r;
      mOut [(size_t)gRow * P_PARTS + p] = mrun[r];
      e1Out[(size_t)gRow * P_PARTS + p] = e1run[r];
      e2Out[(size_t)gRow * P_PARTS + p] = e2run[r];
    }
  }
  if (mode != 4) {
    lsum = wave_reduce_add(lsum);
    if (lane == 0) atomicAdd(&accum[accSlot], lsum);
  }
}

// Merge column-partials, compute per-row dcm / dtm / hinge terms.
__global__ void combine_k(const float* __restrict__ pos, const float* __restrict__ dDiag,
                          const float* __restrict__ gtModal,
                          const float* __restrict__ ngM, const float* __restrict__ ngE1,
                          const float* __restrict__ ngE2,
                          const float* __restrict__ emM, const float* __restrict__ emE1,
                          const float* __restrict__ emE2,
                          float* __restrict__ accum) {
  const int i = blockIdx.x * blockDim.x + threadIdx.x;
  float dcm_i = 0.f, dtm_i = 0.f, rank_i = 0.f;
  const float invT   = 1.0f / T_CONST;
  const float denom  = 1.0f - TAU_CONST + 1e-12f;
  const float floorv = expf(-1.0f / T_CONST);
  const float Nf     = (float)NG_COLS;

  if (i < B_ROWS) {
    // ---- merge ng stats (m, E1=sum exp((x-m)/T), E2=sum exp(2(x-m)/T)) ----
    float m = -INFINITY, e1 = 0.f, e2 = 0.f;
    for (int p = 0; p < P_PARTS; ++p) {
      const float tm = ngM[(size_t)i * P_PARTS + p];
      const float nm = fmaxf(m, tm);
      const float s1 = expf((m - nm) * invT);
      const float s2 = expf((tm - nm) * invT);
      e1 = e1 * s1 + ngE1[(size_t)i * P_PARTS + p] * s2;
      e2 = e2 * (s1 * s1) + ngE2[(size_t)i * P_PARTS + p] * (s2 * s2);
      m = nm;
    }
    const float posv = pos[i];
    {
      const float mx   = fmaxf(posv, m);
      const float s    = expf((m - mx) * invT);
      const float sum1 = e1 * s;
      const float sum2 = e2 * s * s;
      const float pexp = expf((posv - mx) * invT);
      const float wsum = BETA_CONST * sum2 / (sum1 / Nf + 1e-12f);
      float Ng = (-Nf * TAU_CONST * pexp + wsum) / denom;
      Ng = fmaxf(Ng, floorv);
      dcm_i = -logf(pexp / (pexp + Ng + 1e-12f));
    }
    // ---- merge emb stats ----
    float m2 = -INFINITY, f1 = 0.f, f2 = 0.f;
    for (int p = 0; p < P_PARTS; ++p) {
      const float tm = emM[(size_t)i * P_PARTS + p];
      const float nm = fmaxf(m2, tm);
      const float s1 = expf((m2 - nm) * invT);
      const float s2 = expf((tm - nm) * invT);
      f1 = f1 * s1 + emE1[(size_t)i * P_PARTS + p] * s2;
      f2 = f2 * (s1 * s1) + emE2[(size_t)i * P_PARTS + p] * (s2 * s2);
      m2 = nm;
    }
    const float wneg = BETA_CONST * f2 / (f1 / Nf + 1e-12f);

    // ---- positives: clean + 5 noisy (JAX threefry noise, key=42) ----
    float ap[6];
    ap[0] = gtModal[i] * invT;
    #pragma unroll
    for (int j = 0; j < 5; ++j) {
      const unsigned idx = (unsigned)(i * 5 + j);
      unsigned b0, b1, bits;
      if (idx < 5120u) { threefry2x32_0_42(idx, idx + 5120u, b0, b1); bits = b0; }
      else             { threefry2x32_0_42(idx - 5120u, idx, b0, b1); bits = b1; }
      const float nz = bits_to_normal(bits);
      ap[1 + j] = (posv + LAM_CONST * nz) * invT;
    }
    float rm = ap[0];
    #pragma unroll
    for (int j = 1; j < 6; ++j) rm = fmaxf(rm, ap[j]);
    float pes = 0.f, pe2 = 0.f;
    #pragma unroll
    for (int j = 0; j < 6; ++j) { const float e = expf(ap[j] - rm); pes += e; pe2 += e * e; }
    const float wpos = BETA_CONST * pe2 / (pes / 6.0f + 1e-12f);
    float hn = (wneg - Nf * TAU_CONST * wpos) / denom;
    hn = fmaxf(hn, floorv);
    dtm_i = -logf(pes / (pes + hn + 1e-12f));

    rank_i = fmaxf(0.0f, posv - dDiag[i] + MARGIN_CONST);
  }

  dcm_i  = wave_reduce_add(dcm_i);
  dtm_i  = wave_reduce_add(dtm_i);
  rank_i = wave_reduce_add(rank_i);
  if ((threadIdx.x & 31) == 0) {
    atomicAdd(&accum[3], dcm_i);
    atomicAdd(&accum[4], dtm_i);
    atomicAdd(&accum[5], rank_i);
  }
}

__global__ void final_k(const float* __restrict__ accum, float* __restrict__ out) {
  if (threadIdx.x == 0) {
    const float Bf = (float)B_ROWS;
    const float siglip  = (accum[0] + accum[1]) / Bf;  // gt + ng-negatives
    const float dense_c = accum[2] / Bf;
    const float dcm     = accum[3] / Bf;
    const float dtm     = accum[4] / Bf;
    const float rank    = accum[5] / Bf;
    out[0] = siglip + W_DCM * dcm + W_DTM * dtm + W_DENSE * (dense_c + rank);
  }
}

// ---------------------------------------------------------------- launcher

extern "C" void kernel_launch(void* const* d_in, const int* in_sizes, int n_in,
                              void* d_out, int out_size, void* d_ws, size_t ws_size,
                              hipStream_t stream) {
  (void)in_sizes; (void)n_in; (void)out_size; (void)ws_size;
  const float* img = (const float*)d_in[0];   // (2048, 1024) f32
  const float* txt = (const float*)d_in[1];   // (20480, 1024) f32
  const float* ls  = (const float*)d_in[2];   // scalar
  const float* lb  = (const float*)d_in[3];   // scalar

  char* base = (char*)d_ws;
  size_t off = 0;
  auto bump = [&](size_t bytes) -> char* {
    off = (off + 255) & ~(size_t)255;
    char* p = base + off;
    off += bytes;
    return p;
  };

  float* accum   = (float*)bump(64 * sizeof(float));
  float* pos     = (float*)bump((size_t)B_ROWS * sizeof(float));
  float* dDiag   = (float*)bump((size_t)B_ROWS * sizeof(float));
  float* gtModal = (float*)bump((size_t)B_ROWS * sizeof(float));
  float* ngM  = (float*)bump((size_t)B_ROWS * P_PARTS * sizeof(float));
  float* ngE1 = (float*)bump((size_t)B_ROWS * P_PARTS * sizeof(float));
  float* ngE2 = (float*)bump((size_t)B_ROWS * P_PARTS * sizeof(float));
  float* emM  = (float*)bump((size_t)B_ROWS * P_PARTS * sizeof(float));
  float* emE1 = (float*)bump((size_t)B_ROWS * P_PARTS * sizeof(float));
  float* emE2 = (float*)bump((size_t)B_ROWS * P_PARTS * sizeof(float));
  _Float16* imgHi = (_Float16*)bump((size_t)B_ROWS * D_DIM * sizeof(_Float16));
  _Float16* imgLo = (_Float16*)bump((size_t)B_ROWS * D_DIM * sizeof(_Float16));
  _Float16* txtHi = (_Float16*)bump((size_t)10 * B_ROWS * D_DIM * sizeof(_Float16));
  _Float16* txtLo = (_Float16*)bump((size_t)10 * B_ROWS * D_DIM * sizeof(_Float16));

  const size_t gtOff    = 0;
  const size_t denseOff = (size_t)B_ROWS * D_DIM;
  const size_t ngOff    = (size_t)2 * B_ROWS * D_DIM;

  const size_t shmem = (size_t)64 * D_DIM * 2 /*planes*/ * sizeof(_Float16); // 256 KB
  hipFuncSetAttribute((const void*)gemm_fused_k,
                      hipFuncAttributeMaxDynamicSharedMemorySize, (int)shmem);

  init_accum_k<<<1, 64, 0, stream>>>(accum);
  split_f16_k<<<2048, 256, 0, stream>>>(img, imgHi, imgLo, (size_t)B_ROWS * D_DIM);
  split_f16_k<<<4096, 256, 0, stream>>>(txt, txtHi, txtLo, (size_t)10 * B_ROWS * D_DIM);

  row_dot_k<<<B_ROWS / 8, 256, 0, stream>>>(img, txt + gtOff,    ls, pos);     // diag(img@gt^T)
  row_dot_k<<<B_ROWS / 8, 256, 0, stream>>>(img, txt + denseOff, ls, dDiag);   // diag(img@dense^T)
  row_dot_k<<<B_ROWS / 8, 256, 0, stream>>>(txt + gtOff, txt + gtOff, ls, gtModal); // ||gt||^2

  dim3 grid(B_ROWS / 64, 8), block(256);
  // mode 1: img @ gt^T (2048 cols -> 256 per col-block), siglip sum -> accum[0]
  gemm_fused_k<<<grid, block, shmem, stream>>>(imgHi, imgLo, txtHi + gtOff, txtLo + gtOff,
                                               ls, lb, 256, 1,
                                               nullptr, nullptr, nullptr, accum, 0);
  // mode 2: img @ dense^T, siglip sum -> accum[2]
  gemm_fused_k<<<grid, block, shmem, stream>>>(imgHi, imgLo, txtHi + denseOff, txtLo + denseOff,
                                               ls, lb, 256, 2,
                                               nullptr, nullptr, nullptr, accum, 2);
  // mode 3: img @ ng^T (16384 cols -> 2048 per col-block), neg siglip -> accum[1], ng stats
  gemm_fused_k<<<grid, block, shmem, stream>>>(imgHi, imgLo, txtHi + ngOff, txtLo + ngOff,
                                               ls, lb, 2048, 3,
                                               ngM, ngE1, ngE2, accum, 1);
  // mode 4: gt @ ng^T, emb stats only
  gemm_fused_k<<<grid, block, shmem, stream>>>(txtHi + gtOff, txtLo + gtOff,
                                               txtHi + ngOff, txtLo + ngOff,
                                               ls, lb, 2048, 4,
                                               emM, emE1, emE2, accum, -1);

  combine_k<<<B_ROWS / 256, 256, 0, stream>>>(pos, dDiag, gtModal,
                                              ngM, ngE1, ngE2, emM, emE1, emE2, accum);
  final_k<<<1, 1, 0, stream>>>(accum, (float*)d_out);
}
